// MashKLAutoEncoder_53644141527734
// MI455X (gfx1250) — matrix operand order repaired and checked by reading, not compiled, not run on hardware
//
#include <hip/hip_runtime.h>
#include <hip/hip_bf16.h>
#include <math.h>

// ---------------------------------------------------------------------------
// MashKLAutoEncoder on MI455X (gfx1250, wave32, WMMA)
// GEMMs: double-buffered ASYNC global->LDS tile staging
// (global_load_async_to_lds_b128 / s_wait_asynccnt) + v_wmma_f32_16x16x32_bf16.
// bf16 conversion happens at fragment build (ds_load_b64 -> v_cvt_pk_bf16_f32).
// ---------------------------------------------------------------------------

#define DEV __device__ __forceinline__
#define CEILDIV(a, b) (((a) + (b) - 1) / (b))

typedef __bf16 bf16;
typedef __attribute__((ext_vector_type(16))) __bf16 v16bf;
typedef __attribute__((ext_vector_type(8))) float v8f;

DEV float wred_sum(float v) {
#pragma unroll
  for (int i = 16; i > 0; i >>= 1) v += __shfl_xor(v, i, 32);
  return v;
}
DEV float wred_max(float v) {
#pragma unroll
  for (int i = 16; i > 0; i >>= 1) v = fmaxf(v, __shfl_xor(v, i, 32));
  return v;
}
DEV float sigmoid_f(float x) { return 1.f / (1.f + expf(-x)); }
DEV float gelu_tanh(float g) {
  float t = tanhf(0.7978845608028654f * (g + 0.044715f * g * g * g));
  return 0.5f * g * (1.f + t);
}
DEV unsigned short f2bf_bits(float x) {
  union { bf16 b; unsigned short s; } u;
  u.b = (bf16)x;
  return u.s;
}
DEV unsigned pack2(float lo, float hi) {
  return (unsigned)f2bf_bits(lo) | ((unsigned)f2bf_bits(hi) << 16);
}

// Async copy of 16 bytes global -> LDS (GV mode, ASYNCcnt-tracked).
DEV void async_ld_b128(void* lds_dst, const void* gsrc) {
  unsigned off = (unsigned)(unsigned long long)lds_dst;  // low 32 bits = LDS offset
  asm volatile("global_load_async_to_lds_b128 %0, %1, off" ::"v"(off), "v"(gsrc)
               : "memory");
}
DEV void wait_async0() { asm volatile("s_wait_asynccnt 0x0" ::: "memory"); }

// ---------------------------------------------------------------------------
// Batched tiled GEMM.  Block = 128 threads (4 waves) -> 64x64 C tile.
//   TRANSB=true :  C[m,n] = sum_k A[m,k] * B[n,k]   (x @ w.T)
//   TRANSB=false:  C[m,n] = sum_k A[m,k] * B[k,n]
// A LDS tile:  f32 [64 rows][36]   (stride mult-of-4 -> async B128 aligned)
// B LDS tile:  TRANSB -> f32 [64 n][36 k] (no transpose needed for async)
//              !TRANSB-> f32 [32 k][68 n]
// Double buffered; steady state: issue async copies for K-step k+1, run 4
// WMMAs on K-step k, s_wait_asynccnt 0 + barrier, swap.  Out-of-range rows/
// cols are zero-filled at staging so fragments need no selects.  K-tails use
// a clamped synchronous path into the same layout.
// ---------------------------------------------------------------------------
template <bool TRANSB>
__global__ void k_gemm_wmma(const float* __restrict__ A, const float* __restrict__ B,
                            const float* __restrict__ bias, float* __restrict__ C,
                            int M, int N, int K, int lda, int ldb, int ldc, int H,
                            long long sAb, long long sAh, long long sBb, long long sBh,
                            long long sCb, long long sCh) {
  constexpr int BSIZE = TRANSB ? (64 * 36) : (32 * 68);
  __shared__ float Asf[2][64 * 36];
  __shared__ float Bsf[2][BSIZE];

  int z = blockIdx.z;
  int bz = z / H, hz = z % H;
  A += (long long)bz * sAb + (long long)hz * sAh;
  B += (long long)bz * sBb + (long long)hz * sBh;
  C += (long long)bz * sCb + (long long)hz * sCh;

  const int tid = threadIdx.x;
  const int wave = tid >> 5;
  const int lane = tid & 31;
  const int row0 = blockIdx.x * 64;
  const int col0 = blockIdx.y * 64;
  const bool lda_vec = ((lda & 3) == 0);
  const bool ldb_vec = ((ldb & 3) == 0);

  v8f acc[4] = {};

  auto stageA = [&](int bufi, int k0) {
    const bool fast = (k0 + 32 <= K) && lda_vec;
#pragma unroll
    for (int i = 0; i < 4; i++) {
      int idx = tid + i * 128;
      int r = idx >> 3;
      int kq = (idx & 7) * 4;
      int grow = row0 + r;
      float* dst = &Asf[bufi][r * 36 + kq];
      if (fast) {
        if (grow < M) {
          async_ld_b128(dst, A + (long long)grow * lda + k0 + kq);
        } else {
          *(float4*)dst = float4{0.f, 0.f, 0.f, 0.f};
        }
      } else {
        int grc = grow < M ? grow : M - 1;
        const float* ap = A + (long long)grc * lda;
        int k1 = k0 + kq;
        bool rok = grow < M;
        float v0 = ap[(k1 + 0) < K ? (k1 + 0) : (K - 1)];
        float v1 = ap[(k1 + 1) < K ? (k1 + 1) : (K - 1)];
        float v2 = ap[(k1 + 2) < K ? (k1 + 2) : (K - 1)];
        float v3 = ap[(k1 + 3) < K ? (k1 + 3) : (K - 1)];
        dst[0] = (rok && (k1 + 0) < K) ? v0 : 0.f;
        dst[1] = (rok && (k1 + 1) < K) ? v1 : 0.f;
        dst[2] = (rok && (k1 + 2) < K) ? v2 : 0.f;
        dst[3] = (rok && (k1 + 3) < K) ? v3 : 0.f;
      }
    }
  };

  auto stageB = [&](int bufi, int k0) {
    const bool fast = (k0 + 32 <= K) && ldb_vec;
#pragma unroll
    for (int i = 0; i < 4; i++) {
      int idx = tid + i * 128;
      if (TRANSB) {
        int n0 = idx >> 3;       // B^T column (output col)
        int kq = (idx & 7) * 4;  // along K
        int gn = col0 + n0;
        float* dst = &Bsf[bufi][n0 * 36 + kq];
        if (fast) {
          if (gn < N) {
            async_ld_b128(dst, B + (long long)gn * ldb + k0 + kq);
          } else {
            *(float4*)dst = float4{0.f, 0.f, 0.f, 0.f};
          }
        } else {
          int gnc = gn < N ? gn : N - 1;
          const float* bp = B + (long long)gnc * ldb;
          int k1 = k0 + kq;
          bool nok = gn < N;
          float v0 = bp[(k1 + 0) < K ? (k1 + 0) : (K - 1)];
          float v1 = bp[(k1 + 1) < K ? (k1 + 1) : (K - 1)];
          float v2 = bp[(k1 + 2) < K ? (k1 + 2) : (K - 1)];
          float v3 = bp[(k1 + 3) < K ? (k1 + 3) : (K - 1)];
          dst[0] = (nok && (k1 + 0) < K) ? v0 : 0.f;
          dst[1] = (nok && (k1 + 1) < K) ? v1 : 0.f;
          dst[2] = (nok && (k1 + 2) < K) ? v2 : 0.f;
          dst[3] = (nok && (k1 + 3) < K) ? v3 : 0.f;
        }
      } else {
        int kk = idx >> 4;        // K row
        int n0 = (idx & 15) * 4;  // along N (N assumed mult of 4 on this path)
        int gk = k0 + kk;
        float* dst = &Bsf[bufi][kk * 68 + n0];
        if (fast) {
          if ((col0 + n0) < N) {  // cols stay inside row (N <= ldb)
            async_ld_b128(dst, B + (long long)gk * ldb + col0 + n0);
          } else {
            *(float4*)dst = float4{0.f, 0.f, 0.f, 0.f};
          }
        } else {
          int gkc = gk < K ? gk : K - 1;
          const float* bp = B + (long long)gkc * ldb + col0;
          bool kok = gk < K;
          float v0 = bp[n0 + 0], v1 = bp[n0 + 1], v2 = bp[n0 + 2], v3 = bp[n0 + 3];
          dst[0] = (kok && (col0 + n0 + 0) < N) ? v0 : 0.f;
          dst[1] = (kok && (col0 + n0 + 1) < N) ? v1 : 0.f;
          dst[2] = (kok && (col0 + n0 + 2) < N) ? v2 : 0.f;
          dst[3] = (kok && (col0 + n0 + 3) < N) ? v3 : 0.f;
        }
      }
    }
  };

  // Prologue: stage first K tile.
  stageA(0, 0);
  stageB(0, 0);
  wait_async0();
  __syncthreads();

  int buf = 0;
  for (int k0 = 0; k0 < K; k0 += 32) {
    const bool more = (k0 + 32) < K;
    if (more) {  // issue async copies for next tile into the other buffer
      stageA(buf ^ 1, k0 + 32);
      stageB(buf ^ 1, k0 + 32);
    }

    // ---------------- fragments + 4 WMMAs per wave --------------------------
    union FragU { v16bf v; unsigned u[8]; };
    FragU fa[4], fb;
    const int khalf = lane >> 4;
    const int rsub = lane & 15;
#pragma unroll
    for (int v = 0; v < 8; v++) {
      int kk = (v < 4) ? (khalf * 8 + 2 * v) : (16 + khalf * 8 + 2 * (v - 4));
      if (TRANSB) {
        int n0 = wave * 16 + rsub;
        float b0 = Bsf[buf][n0 * 36 + kk + 0];
        float b1 = Bsf[buf][n0 * 36 + kk + 1];
        fb.u[v] = pack2(b0, b1);
      } else {
        float b0 = Bsf[buf][(kk + 0) * 68 + wave * 16 + rsub];
        float b1 = Bsf[buf][(kk + 1) * 68 + wave * 16 + rsub];
        fb.u[v] = pack2(b0, b1);
      }
#pragma unroll
      for (int s = 0; s < 4; s++) {
        float2 a2 = *(const float2*)&Asf[buf][(s * 16 + rsub) * 36 + kk];
        fa[s].u[v] = pack2(a2.x, a2.y);
      }
    }
#pragma unroll
    for (int s = 0; s < 4; s++)
      acc[s] = __builtin_amdgcn_wmma_f32_16x16x32_bf16(false, fa[s].v, false, fb.v,
                                                       (short)0, acc[s], false, false);

    wait_async0();   // next tile's async copies have landed
    __syncthreads(); // everyone done reading buf / writing buf^1
    buf ^= 1;
  }

  // ---------------- store ---------------------------------------------------
  int cn = col0 + wave * 16 + (lane & 15);
  float bb = (bias != nullptr && cn < N) ? bias[cn] : 0.f;
#pragma unroll
  for (int s = 0; s < 4; s++) {
#pragma unroll
    for (int v = 0; v < 8; v++) {
      int cm = row0 + s * 16 + v + 8 * (lane >> 4);
      if (cm < M && cn < N) C[(long long)cm * ldc + cn] = acc[s][v] + bb;
    }
  }
}

// ---------------------------------------------------------------------------
// Fourier point-embed: one token per block; features in LDS, then 100 dots.
// ---------------------------------------------------------------------------
__global__ void k_point_embed(const float* __restrict__ mash, const float* __restrict__ w,
                              const float* __restrict__ bias, float* __restrict__ out,
                              int c, int nf, int ch_off, int col_off) {
  __shared__ float feat[128];
  int token = blockIdx.x;
  const float* x = mash + (long long)token * 22 + ch_off;
  int F = 2 * c * nf + c;
  for (int i = threadIdx.x; i < c * nf; i += blockDim.x) {
    int ch = i / nf, j = i % nf;
    float p = x[ch] * exp2f((float)j) * 3.14159265358979323846f;
    feat[i] = sinf(p);
    feat[c * nf + i] = cosf(p);
  }
  for (int i = threadIdx.x; i < c; i += blockDim.x) feat[2 * c * nf + i] = x[i];
  __syncthreads();
  for (int j = threadIdx.x; j < 100; j += blockDim.x) {
    float acc = bias[j];
    const float* wr = w + (long long)j * F;
    for (int f = 0; f < F; f++) acc = fmaf(feat[f], wr[f], acc);
    out[(long long)token * 400 + col_off + j] = acc;
  }
}

// ---------------------------------------------------------------------------
// Wave-per-row LayerNorm / fused add+RMSNorm / softmax / final projection.
// ---------------------------------------------------------------------------
__global__ void k_layernorm(const float* __restrict__ in, float* __restrict__ out,
                            const float* __restrict__ w, const float* __restrict__ b,
                            int rows, int dim) {
  int row = blockIdx.x * (blockDim.x >> 5) + (threadIdx.x >> 5);
  int lane = threadIdx.x & 31;
  if (row >= rows) return;
  const float* r = in + (long long)row * dim;
  float s = 0.f, s2 = 0.f;
  for (int i = lane; i < dim; i += 32) { float v = r[i]; s += v; s2 += v * v; }
  s = wred_sum(s); s2 = wred_sum(s2);
  float mean = s / dim;
  float var = s2 / dim - mean * mean;
  float rs = rsqrtf(var + 1e-5f);
  float* o = out + (long long)row * dim;
  for (int i = lane; i < dim; i += 32) o[i] = (r[i] - mean) * rs * w[i] + b[i];
}

__global__ void k_add_rmsnorm(const float* __restrict__ h, const float* __restrict__ rr,
                              const float* __restrict__ w, float* __restrict__ res_out,
                              float* __restrict__ x_out, int rows, int dim) {
  int row = blockIdx.x * (blockDim.x >> 5) + (threadIdx.x >> 5);
  int lane = threadIdx.x & 31;
  if (row >= rows) return;
  long long base = (long long)row * dim;
  float s2 = 0.f;
  for (int i = lane; i < dim; i += 32) {
    float v = h[base + i] + rr[base + i];
    res_out[base + i] = v;
    s2 += v * v;
  }
  s2 = wred_sum(s2);
  float rs = rsqrtf(s2 / dim + 1e-5f);
  for (int i = lane; i < dim; i += 32) x_out[base + i] = res_out[base + i] * rs * w[i];
}

__global__ void k_softmax(float* __restrict__ s, int rows, int len, float scale) {
  int row = blockIdx.x * (blockDim.x >> 5) + (threadIdx.x >> 5);
  int lane = threadIdx.x & 31;
  if (row >= rows) return;
  float* r = s + (long long)row * len;
  float mx = -3.4e38f;
  for (int i = lane; i < len; i += 32) mx = fmaxf(mx, r[i] * scale);
  mx = wred_max(mx);
  float sum = 0.f;
  for (int i = lane; i < len; i += 32) { float e = expf(r[i] * scale - mx); r[i] = e; sum += e; }
  sum = wred_sum(sum);
  float inv = 1.f / sum;
  for (int i = lane; i < len; i += 32) r[i] *= inv;
}

__global__ void k_final_out(const float* __restrict__ lat, const float* __restrict__ ow,
                            const float* __restrict__ ob, float* __restrict__ out, int rows) {
  int row = blockIdx.x * (blockDim.x >> 5) + (threadIdx.x >> 5);
  int lane = threadIdx.x & 31;
  if (row >= rows) return;
  float acc = 0.f;
  for (int i = lane; i < 400; i += 32) acc = fmaf(lat[(long long)row * 400 + i], ow[i], acc);
  acc = wred_sum(acc);
  if (lane == 0) out[row] = acc + ob[0];
}

// ---------------------------------------------------------------------------
// Elementwise helpers.
// ---------------------------------------------------------------------------
__global__ void k_zero(float* p, int n) {
  int i = blockIdx.x * blockDim.x + threadIdx.x;
  if (i < n) p[i] = 0.f;
}
__global__ void k_add_inplace(float* __restrict__ d, const float* __restrict__ s, int n) {
  int i = blockIdx.x * blockDim.x + threadIdx.x;
  if (i < n) d[i] += s[i];
}
__global__ void k_geglu(const float* __restrict__ h, float* __restrict__ out, int rows, int half) {
  int i = blockIdx.x * blockDim.x + threadIdx.x;
  if (i >= rows * half) return;
  int r = i / half, j = i % half;
  float a = h[(long long)r * 2 * half + j];
  float g = h[(long long)r * 2 * half + half + j];
  out[i] = a * gelu_tanh(g);
}
__global__ void k_zkl(const float* __restrict__ mean, const float* __restrict__ lvin,
                      const float* __restrict__ eps, float* __restrict__ z,
                      float* __restrict__ kl, int n) {
  int i = blockIdx.x * blockDim.x + threadIdx.x;
  float e = 0.f;
  int b = 0;
  if (i < n) {
    float m = mean[i];
    float lv = fminf(fmaxf(lvin[i], -30.f), 20.f);
    z[i] = m + expf(0.5f * lv) * eps[i];
    e = 0.5f * (m * m + expf(lv) - 1.f - lv);
    b = i / (400 * 64);  // 25600 elems per batch, wave never straddles batches
  }
  e = wred_sum(e);
  if ((threadIdx.x & 31) == 0 && i < n) atomicAdd(&kl[b], e);
}
__global__ void k_conv_silu(const float* __restrict__ xz, const float* __restrict__ cw,
                            const float* __restrict__ cb, float* __restrict__ xc, int total) {
  int i = blockIdx.x * blockDim.x + threadIdx.x;
  if (i >= total) return;
  int r = i / 800, d = i % 800;
  int t = r % 400;
  float acc = cb[d];
#pragma unroll
  for (int k = 0; k < 4; k++) {
    int tt = t + k - 3;
    if (tt >= 0) acc = fmaf(xz[(long long)(r + k - 3) * 1600 + d], cw[d * 4 + k], acc);
  }
  xc[i] = acc * sigmoid_f(acc);
}
__global__ void k_softplus(float* __restrict__ p, int n) {
  int i = blockIdx.x * blockDim.x + threadIdx.x;
  if (i >= n) return;
  float x = p[i];
  p[i] = (x > 20.f) ? x : log1pf(expf(x));
}
__global__ void k_ssm_out(float* __restrict__ ys, const float* __restrict__ xc,
                          const float* __restrict__ Dp, const float* __restrict__ xz, int total) {
  int i = blockIdx.x * blockDim.x + threadIdx.x;
  if (i >= total) return;
  int r = i / 800, d = i % 800;
  float zg = xz[(long long)r * 1600 + 800 + d];
  ys[i] = (ys[i] + Dp[d] * xc[i]) * (zg * sigmoid_f(zg));
}

// ---------------------------------------------------------------------------
// Selective-scan: one (b,d) channel per lane, 16 states in VGPRs, 400 steps.
// dbc rows padded to 64 floats (cols: dt 0..24, B 25..40, C 41..56); the row
// pointer is block-uniform per timestep -> scalar-load friendly.
// ---------------------------------------------------------------------------
__global__ void k_ssm_scan(const float* __restrict__ dt, const float* __restrict__ xc,
                           const float* __restrict__ dbc, const float* __restrict__ Alog,
                           float* __restrict__ ys, int L) {
  int d = blockIdx.x * blockDim.x + threadIdx.x;
  int b = blockIdx.y;
  if (d >= 800) return;
  float A[16], h[16];
#pragma unroll
  for (int n = 0; n < 16; n++) {
    A[n] = -expf(Alog[(long long)d * 16 + n]);
    h[n] = 0.f;
  }
  for (int t = 0; t < L; t++) {
    long long rb = (long long)(b * L + t);
    float dtv = dt[rb * 800 + d];
    float xv = xc[rb * 800 + d];
    const float* bc = dbc + rb * 64 + 25;  // Bm[0:16], Cm[16:32]
    float dtx = dtv * xv;
    float y = 0.f;
#pragma unroll
    for (int n = 0; n < 16; n++) {
      float dA = expf(dtv * A[n]);
      h[n] = fmaf(dA, h[n], dtx * bc[n]);
      y = fmaf(h[n], bc[16 + n], y);
    }
    ys[rb * 800 + d] = y;
  }
}

// ---------------------------------------------------------------------------
// Host orchestration.
// ---------------------------------------------------------------------------
static void gemm(bool transb, const float* A, const float* B, const float* bias, float* C,
                 int M, int N, int K, int lda, int ldb, int ldc, int Z, int H,
                 long long sAb, long long sAh, long long sBb, long long sBh,
                 long long sCb, long long sCh, hipStream_t s) {
  dim3 g(CEILDIV(M, 64), CEILDIV(N, 64), Z), blk(128);
  if (transb)
    k_gemm_wmma<true><<<g, blk, 0, s>>>(A, B, bias, C, M, N, K, lda, ldb, ldc, H,
                                        sAb, sAh, sBb, sBh, sCb, sCh);
  else
    k_gemm_wmma<false><<<g, blk, 0, s>>>(A, B, bias, C, M, N, K, lda, ldb, ldc, H,
                                         sAb, sAh, sBb, sBh, sCb, sCh);
}
static void gemm1(bool transb, const float* A, const float* B, const float* bias, float* C,
                  int M, int N, int K, int lda, int ldb, int ldc, hipStream_t s) {
  gemm(transb, A, B, bias, C, M, N, K, lda, ldb, ldc, 1, 1, 0, 0, 0, 0, 0, 0, s);
}

extern "C" void kernel_launch(void* const* d_in, const int* in_sizes, int n_in,
                              void* d_out, int out_size, void* d_ws, size_t ws_size,
                              hipStream_t stream) {
  (void)in_sizes; (void)n_in; (void)out_size; (void)ws_size;
  const float* P[57];
  for (int i = 0; i < 57; i++) P[i] = (const float*)d_in[i];

  const float* mash = P[0];
  const float* eps = P[1];
  float* out = (float*)d_out;  // [1600] lat, then [4] kl

  // ---- workspace arena (floats) ----
  float* W = (float*)d_ws;
  long long off = 0;
  auto alloc = [&](long long n) { float* p = W + off; off += n; return p; };
  float* o_h    = alloc(640000);    // hidden / emb / lat stream
  float* o_res  = alloc(640000);    // mamba residual
  float* o_x    = alloc(640000);    // normed activations
  float* o_lnc  = alloc(640000);    // second layernorm
  float* o_big1 = alloc(5120000);   // kv / ff hidden
  float* o_big2 = alloc(2560000);   // q / xz
  float* o_big3 = alloc(2560000);   // scores / geglu inner
  float* o_big4 = alloc(2560000);   // attention output
  float* o_xc   = alloc(1280000);
  float* o_dt   = alloc(1280000);
  float* o_ys   = alloc(1280000);
  float* o_dbc  = alloc(102400);    // [1600,64] (57 used, padded for alignment)
  float* o_mean = alloc(102400);
  float* o_lv   = alloc(102400);
  float* o_z    = alloc(102400);

  dim3 b128(128), b256(256);
  auto rows4 = [](int rows) { return dim3(CEILDIV(rows, 4)); };
  const float ATT_SCALE = 1.f / 20.f;  // dh=400 -> dh^-0.5

  // ================= Encoder =================
  k_point_embed<<<1600, b128, 0, stream>>>(mash, P[2], P[3], o_h, 3, 16, 0, 0);
  k_point_embed<<<1600, b128, 0, stream>>>(mash, P[4], P[5], o_h, 3, 16, 3, 100);
  k_point_embed<<<1600, b128, 0, stream>>>(mash, P[6], P[7], o_h, 7, 7, 6, 200);
  k_point_embed<<<1600, b128, 0, stream>>>(mash, P[8], P[9], o_h, 9, 5, 13, 300);

  k_layernorm<<<rows4(1600), b128, 0, stream>>>(o_h, o_x, P[10], P[11], 1600, 400);
  k_layernorm<<<rows4(1600), b128, 0, stream>>>(o_h, o_lnc, P[12], P[13], 1600, 400);
  gemm1(true, o_x, P[14], nullptr, o_big2, 1600, 400, 400, 400, 400, 400, stream);   // q
  gemm1(true, o_lnc, P[15], nullptr, o_big1, 1600, 800, 400, 400, 400, 800, stream); // kv
  gemm(true, o_big2, o_big1, nullptr, o_big3, 400, 400, 400, 400, 800, 400, 4, 1,
       160000, 0, 320000, 0, 160000, 0, stream);                                     // scores
  k_softmax<<<rows4(1600), b128, 0, stream>>>(o_big3, 1600, 400, ATT_SCALE);
  gemm(false, o_big3, o_big1 + 400, nullptr, o_big4, 400, 400, 400, 400, 800, 400, 4, 1,
       160000, 0, 320000, 0, 160000, 0, stream);                                     // att @ v
  gemm1(true, o_big4, P[16], P[17], o_x, 1600, 400, 400, 400, 400, 400, stream);     // o proj
  k_add_inplace<<<CEILDIV(640000, 256), b256, 0, stream>>>(o_h, o_x, 640000);        // + emb

  k_layernorm<<<rows4(1600), b128, 0, stream>>>(o_h, o_x, P[18], P[19], 1600, 400);
  gemm1(true, o_x, P[20], P[21], o_big1, 1600, 3200, 400, 400, 400, 3200, stream);
  k_geglu<<<CEILDIV(2560000, 256), b256, 0, stream>>>(o_big1, o_big3, 1600, 1600);
  gemm1(true, o_big3, P[22], P[23], o_x, 1600, 400, 1600, 1600, 1600, 400, stream);
  k_add_inplace<<<CEILDIV(640000, 256), b256, 0, stream>>>(o_h, o_x, 640000);

  gemm1(true, o_h, P[24], P[25], o_mean, 1600, 64, 400, 400, 400, 64, stream);
  gemm1(true, o_h, P[26], P[27], o_lv, 1600, 64, 400, 400, 400, 64, stream);
  k_zero<<<1, 32, 0, stream>>>(out + 1600, 4);
  k_zkl<<<CEILDIV(102400, 256), b256, 0, stream>>>(o_mean, o_lv, eps, o_z, out + 1600, 102400);
  gemm1(true, o_z, P[28], P[29], o_h, 1600, 400, 64, 64, 64, 400, stream);           // hidden
  k_zero<<<CEILDIV(640000, 256), b256, 0, stream>>>(o_res, 640000);

  // ================= 96-layer Mamba stack =================
  for (int l = 0; l < 96; l++) {
    const float* nw = P[30] + (long long)l * 400;
    const float* inw = P[31] + (long long)l * 1600 * 400;
    const float* cw = P[32] + (long long)l * 800 * 4;
    const float* cb = P[33] + (long long)l * 800;
    const float* xw = P[34] + (long long)l * 57 * 800;
    const float* dtw = P[35] + (long long)l * 800 * 25;
    const float* dtb = P[36] + (long long)l * 800;
    const float* alog = P[37] + (long long)l * 800 * 16;
    const float* Dp = P[38] + (long long)l * 800;
    const float* ow = P[39] + (long long)l * 400 * 800;

    k_add_rmsnorm<<<rows4(1600), b128, 0, stream>>>(o_h, o_res, nw, o_res, o_x, 1600, 400);
    gemm1(true, o_x, inw, nullptr, o_big2, 1600, 1600, 400, 400, 400, 1600, stream); // xz
    k_conv_silu<<<CEILDIV(1280000, 256), b256, 0, stream>>>(o_big2, cw, cb, o_xc, 1280000);
    gemm1(true, o_xc, xw, nullptr, o_dbc, 1600, 57, 800, 800, 800, 64, stream);      // dt,B,C
    gemm1(true, o_dbc, dtw, dtb, o_dt, 1600, 800, 25, 64, 25, 800, stream);          // dt proj
    k_softplus<<<CEILDIV(1280000, 256), b256, 0, stream>>>(o_dt, 1280000);
    k_ssm_scan<<<dim3(CEILDIV(800, 256), 4), b256, 0, stream>>>(o_dt, o_xc, o_dbc, alog, o_ys, 400);
    k_ssm_out<<<CEILDIV(1280000, 256), b256, 0, stream>>>(o_ys, o_xc, Dp, o_big2, 1280000);
    gemm1(true, o_ys, ow, nullptr, o_h, 1600, 400, 800, 800, 800, 400, stream);      // out proj
  }

  // ================= Decoder =================
  k_add_rmsnorm<<<rows4(1600), b128, 0, stream>>>(o_h, o_res, P[40], o_res, o_x, 1600, 400);
  k_layernorm<<<rows4(1600), b128, 0, stream>>>(o_x, o_h, P[41], P[42], 1600, 400);    // lnq
  k_layernorm<<<rows4(1600), b128, 0, stream>>>(o_x, o_lnc, P[43], P[44], 1600, 400);  // lnc
  gemm1(true, o_h, P[45], nullptr, o_big2, 1600, 1600, 400, 400, 400, 1600, stream);   // q
  gemm1(true, o_lnc, P[46], nullptr, o_big1, 1600, 3200, 400, 400, 400, 3200, stream); // kv
  // scores[b,h] = q_bh @ k_bh^T  (16 batched problems, head = column slice)
  gemm(true, o_big2, o_big1, nullptr, o_big3, 400, 400, 400, 1600, 3200, 400, 16, 4,
       640000, 400, 1280000, 400, 640000, 160000, stream);
  k_softmax<<<rows4(6400), b128, 0, stream>>>(o_big3, 6400, 400, ATT_SCALE);
  gemm(false, o_big3, o_big1 + 1600, nullptr, o_big4, 400, 400, 400, 400, 3200, 1600, 16, 4,
       640000, 160000, 1280000, 400, 640000, 400, stream);                             // att @ v
  gemm1(true, o_big4, P[47], P[48], o_h, 1600, 400, 1600, 1600, 1600, 400, stream);    // lat

  k_layernorm<<<rows4(1600), b128, 0, stream>>>(o_h, o_x, P[49], P[50], 1600, 400);
  gemm1(true, o_x, P[51], P[52], o_big1, 1600, 3200, 400, 400, 400, 3200, stream);
  k_geglu<<<CEILDIV(2560000, 256), b256, 0, stream>>>(o_big1, o_big3, 1600, 1600);
  gemm1(true, o_big3, P[53], P[54], o_x, 1600, 400, 1600, 1600, 1600, 400, stream);
  k_add_inplace<<<CEILDIV(640000, 256), b256, 0, stream>>>(o_h, o_x, 640000);

  k_final_out<<<rows4(1600), b128, 0, stream>>>(o_h, P[55], P[56], out, 1600);
}
